// XEGNNK_47863115547369
// MI455X (gfx1250) — matrix operand order repaired and checked by reading, not compiled, not run on hardware
//
#include <hip/hip_runtime.h>
#include <hip/hip_bf16.h>
#include <math.h>

// ---------------------------------------------------------------------------
// EGNN layer for MI455X (gfx1250), wave32, WMMA f32_16x16x32_f16 matrix core.
// Round 2: fragment-swizzled weights (coalesced b128 B-frag loads), vectorized
// LDS A-frag loads, 128-bit gather/scatter paths.
// ---------------------------------------------------------------------------

typedef __attribute__((ext_vector_type(16))) _Float16 v16h;
typedef __attribute__((ext_vector_type(8)))  _Float16 v8h;
typedef __attribute__((ext_vector_type(4)))  _Float16 v4h;
typedef __attribute__((ext_vector_type(8)))  float    v8f;
typedef __attribute__((ext_vector_type(4)))  float    v4f;

#define NN 20000
#define EE 200000
#define BB 64
#define KK 128
#define DD 64
#define TT 256
#define MSG_IN 512
#define EPS_E3 1e-5f
#define EPS_LN 1e-5f
#define CLAMP_V 10.0f

// -------------------- weight fragment-swizzle conversion -------------------
// Packs a row-major [Kdim x Ndim] f32 weight into WMMA-B fragment order:
// dst[((nt*ksteps + ks)*32 + lane)*16 + j] = w[(ks*32 + (lane&16?8:0) + (j<8?j:j+8))][nt*16 + (lane&15)]
__global__ void swz_w_kernel(const float* __restrict__ src,
                             _Float16* __restrict__ dst, int Kdim, int Ndim) {
  int idx = blockIdx.x * 256 + threadIdx.x;
  int tot = Kdim * Ndim;
  if (idx >= tot) return;
  int ksteps = Kdim >> 5;
  int j    = idx & 15;
  int lane = (idx >> 4) & 31;
  int t    = idx >> 9;
  int ks   = t % ksteps;
  int nt   = t / ksteps;
  int ncol = nt * 16 + (lane & 15);
  int row  = ks * 32 + ((lane & 16) ? 8 : 0) + (j < 8 ? j : j + 8);
  dst[idx] = (_Float16)src[row * Ndim + ncol];
}

// -------------------- small elementwise / reduction kernels ----------------

__global__ void accum_mean_kernel(const float* __restrict__ X,
                                  const int* __restrict__ batch,
                                  float* __restrict__ sumX,
                                  float* __restrict__ cnt, int n) {
  long long idx = (long long)blockIdx.x * 256 + threadIdx.x;
  long long tot = (long long)n * 384;
  if (idx >= tot) return;
  int i = (int)(idx / 384);
  int j = (int)(idx % 384);
  int b = batch[i];
  atomicAdd(&sumX[b * 384 + j], X[idx]);
  if (j == 0) atomicAdd(&cnt[b], 1.0f);
}

__global__ void center_norm_kernel(const float* __restrict__ X,
                                   const int* __restrict__ batch,
                                   const float* __restrict__ sumX,
                                   const float* __restrict__ cnt,
                                   float* __restrict__ Xn,
                                   float* __restrict__ normsum, int n) {
  long long idx = (long long)blockIdx.x * 256 + threadIdx.x;
  long long tot = (long long)n * KK;
  if (idx >= tot) return;
  int i = (int)(idx / KK);
  int k = (int)(idx % KK);
  int b = batch[i];
  float invc = 1.0f / fmaxf(cnt[b], 1.0f);
  float acc = 0.0f;
#pragma unroll
  for (int d = 0; d < 3; ++d) {
    long long p = ((long long)i * 3 + d) * KK + k;
    float v = X[p] - sumX[(b * 3 + d) * KK + k] * invc;
    acc += v * v;
    Xn[p] = v;
  }
  atomicAdd(&normsum[b * KK + k], sqrtf(acc));
}

__global__ void e3norm_kernel(const int* __restrict__ batch,
                              const float* __restrict__ e3w,
                              const float* __restrict__ normsum,
                              const float* __restrict__ cnt,
                              float* __restrict__ Xn,
                              float* __restrict__ out,
                              float* __restrict__ meanXsum, int n) {
  long long idx = (long long)blockIdx.x * 256 + threadIdx.x;
  long long tot = (long long)n * 384;
  if (idx >= tot) return;
  int i = (int)(idx / 384);
  int j = (int)(idx % 384);
  int k = j % KK;
  int b = batch[i];
  float mn = normsum[b * KK + k] / fmaxf(cnt[b], 1.0f);
  float v = e3w[k] * Xn[idx] / (mn + EPS_E3);
  Xn[idx] = v;
  out[idx] = v;
  atomicAdd(&meanXsum[b * 384 + j], v);
}

__global__ void layernorm_kernel(const float* __restrict__ H,
                                 const float* __restrict__ lnw,
                                 const float* __restrict__ lnb,
                                 float* __restrict__ Hn, int n) {
  int i = blockIdx.x * 256 + threadIdx.x;
  if (i >= n) return;
  const v4f* h4 = (const v4f*)(H + (long long)i * DD);
  float mu = 0.0f;
#pragma unroll
  for (int c = 0; c < DD / 4; ++c) { v4f v = h4[c]; mu += v.x + v.y + v.z + v.w; }
  mu *= (1.0f / DD);
  float var = 0.0f;
#pragma unroll
  for (int c = 0; c < DD / 4; ++c) {
    v4f v = h4[c];
#pragma unroll
    for (int j = 0; j < 4; ++j) { float d = v[j] - mu; var += d * d; }
  }
  var *= (1.0f / DD);
  float rs = rsqrtf(var + EPS_LN);
  v4f* o4 = (v4f*)(Hn + (long long)i * DD);
#pragma unroll
  for (int c = 0; c < DD / 4; ++c) {
    v4f v = h4[c], w = *(const v4f*)(lnw + c * 4), b = *(const v4f*)(lnb + c * 4);
    v4f r;
#pragma unroll
    for (int j = 0; j < 4; ++j) r[j] = (v[j] - mu) * rs * w[j] + b[j];
    o4[c] = r;
  }
}

// -------------------- WMMA fragment helpers (gfx1250 layouts) --------------

// A fragment: 16x32 f16 tile from LDS (row-major, element stride); two
// 16B-aligned ds_load_b128 per lane, no scalar packing.
__device__ inline v16h frag_A(const _Float16* base, int stride, int kOff, int lane) {
  int m = lane & 15;
  int kb = kOff + ((lane & 16) ? 8 : 0);
  const _Float16* p = base + m * stride + kb;
  v8h lo = *(const v8h*)p;
  v8h hi = *(const v8h*)(p + 16);
  return __builtin_shufflevector(lo, hi, 0, 1, 2, 3, 4, 5, 6, 7,
                                 8, 9, 10, 11, 12, 13, 14, 15);
}

// B fragment from fragment-swizzled weights: one 32B contiguous load per lane.
__device__ inline v16h frag_B_swz(const _Float16* base, int ksteps, int ks,
                                  int nt, int lane) {
  return ((const v16h*)base)[(nt * ksteps + ks) * 32 + lane];
}

#define WMMA16(a, b, c) \
  __builtin_amdgcn_wmma_f32_16x16x32_f16(false, (a), false, (b), (short)0, (c), false, false)

__device__ inline float silu_f(float x) { return x / (1.0f + expf(-x)); }

// -------------------- main edge kernel --------------------------------------

#define MSG_S 520   // lds stride (halves): 1040B, 16B-multiple
#define HID_S 72    // lds stride (halves): 144B, 16B-multiple

__global__ void __launch_bounds__(128, 1)
edge_kernel(const float* __restrict__ Xn, const float* __restrict__ Hn,
            const float* __restrict__ te, const int* __restrict__ batch,
            const int* __restrict__ eidx, const float* __restrict__ cnt,
            const float* __restrict__ meanXsum,
            const _Float16* __restrict__ mw1h, const _Float16* __restrict__ mw2h,
            const _Float16* __restrict__ xw1h, const _Float16* __restrict__ xw2h,
            const _Float16* __restrict__ cw1h, const _Float16* __restrict__ cw2h,
            const float* __restrict__ mb1, const float* __restrict__ mb2,
            const float* __restrict__ xb1, const float* __restrict__ xb2,
            const float* __restrict__ cb1, const float* __restrict__ cb2,
            float* __restrict__ out, int nedges) {
  __shared__ __align__(32) _Float16 lds_msg[16 * MSG_S];   // msg_in (f16)
  __shared__ __align__(16) float    lds_rel[16 * 3 * KK];  // rel_coors (f32)
  __shared__ __align__(16) _Float16 lds_h1[16 * HID_S];
  __shared__ __align__(16) _Float16 lds_mij[16 * HID_S];
  __shared__ __align__(16) _Float16 lds_h2[16 * HID_S];
  __shared__ __align__(16) float    lds_cw[16 * KK];
  __shared__ __align__(16) float    lds_ccw[16 * KK];
  __shared__ int s_idx[16], t_idx[16], sb[16];

  const int tid  = threadIdx.x;
  const int lane = tid & 31;
  const int wv   = tid >> 5;
  const int e0   = blockIdx.x * 16;

  if (tid < 16) {
    int eg = e0 + tid;
    if (eg < nedges) {
      int s = eidx[eg];
      int t = eidx[nedges + eg];
      s_idx[tid] = s; t_idx[tid] = t; sb[tid] = batch[s];
    } else {
      s_idx[tid] = -1; t_idx[tid] = -1; sb[tid] = 0;
    }
  }
  __syncthreads();

  // ---- gather msg_in = [H[t], H[s], rel_dist, te[batch[s]]] as f16 --------
  // 4 consecutive columns per thread; chunks never straddle segment borders.
  for (int it = tid * 4; it < 16 * MSG_IN; it += 128 * 4) {
    int e = it >> 9, c = it & 511;
    int s = s_idx[e], t = t_idx[e];
    v4f v = {0.0f, 0.0f, 0.0f, 0.0f};
    if (s >= 0) {
      if (c < 64) {
        v = *(const v4f*)(Hn + (long long)t * DD + c);
      } else if (c < 128) {
        v = *(const v4f*)(Hn + (long long)s * DD + (c - 64));
      } else if (c < 256) {
        int k = c - 128;
        v4f acc = {0.0f, 0.0f, 0.0f, 0.0f};
#pragma unroll
        for (int d = 0; d < 3; ++d) {
          v4f a = *(const v4f*)(Xn + ((long long)s * 3 + d) * KK + k);
          v4f b = *(const v4f*)(Xn + ((long long)t * 3 + d) * KK + k);
          v4f r;
#pragma unroll
          for (int j = 0; j < 4; ++j) { r[j] = a[j] - b[j]; acc[j] += r[j] * r[j]; }
          *(v4f*)&lds_rel[(e * 3 + d) * KK + k] = r;
        }
        v = acc;
      } else {
        v = *(const v4f*)(te + sb[e] * TT + (c - 256));
      }
    }
    v4h h;
#pragma unroll
    for (int j = 0; j < 4; ++j) h[j] = (_Float16)v[j];
    *(v4h*)&lds_msg[e * MSG_S + c] = h;
  }
  __syncthreads();

  const int col = wv * 16 + (lane & 15);
  const int rbase = (lane & 16) ? 8 : 0;

  // ---- GEMM1: msg(16x512) @ mw1(512x64), +mb1, SiLU -> lds_h1 -------------
  {
    v8f acc = {};
#pragma unroll
    for (int ks = 0; ks < 16; ++ks) {
      v16h a = frag_A(lds_msg, MSG_S, ks * 32, lane);
      v16h b = frag_B_swz(mw1h, 16, ks, wv, lane);
      acc = WMMA16(a, b, acc);
    }
    float bias = mb1[col];
#pragma unroll
    for (int r = 0; r < 8; ++r)
      lds_h1[(rbase + r) * HID_S + col] = (_Float16)silu_f(acc[r] + bias);
  }
  __syncthreads();

  // ---- GEMM2: h1(16x64) @ mw2(64x64) + mb2 -> m_ij ------------------------
  {
    v8f acc = {};
#pragma unroll
    for (int ks = 0; ks < 2; ++ks) {
      v16h a = frag_A(lds_h1, HID_S, ks * 32, lane);
      v16h b = frag_B_swz(mw2h, 2, ks, wv, lane);
      acc = WMMA16(a, b, acc);
    }
    float bias = mb2[col];
#pragma unroll
    for (int r = 0; r < 8; ++r)
      lds_mij[(rbase + r) * HID_S + col] = (_Float16)(acc[r] + bias);
  }
  __syncthreads();

  // ---- radial MLP: silu(m_ij@xw1+xb1)@xw2+xb2, clamp -> lds_cw ------------
  {
    v8f acc = {};
#pragma unroll
    for (int ks = 0; ks < 2; ++ks) {
      v16h a = frag_A(lds_mij, HID_S, ks * 32, lane);
      v16h b = frag_B_swz(xw1h, 2, ks, wv, lane);
      acc = WMMA16(a, b, acc);
    }
    float bias = xb1[col];
#pragma unroll
    for (int r = 0; r < 8; ++r)
      lds_h2[(rbase + r) * HID_S + col] = (_Float16)silu_f(acc[r] + bias);
  }
  __syncthreads();
  {
#pragma unroll
    for (int nt = 0; nt < 2; ++nt) {   // each wave covers 2 of 8 N-tiles
      int nTile = wv + nt * 4;
      v8f acc = {};
#pragma unroll
      for (int ks = 0; ks < 2; ++ks) {
        v16h a = frag_A(lds_h2, HID_S, ks * 32, lane);
        v16h b = frag_B_swz(xw2h, 2, ks, nTile, lane);
        acc = WMMA16(a, b, acc);
      }
      int c2 = nTile * 16 + (lane & 15);
      float bias = xb2[c2];
#pragma unroll
      for (int r = 0; r < 8; ++r) {
        float v = fminf(fmaxf(acc[r] + bias, -CLAMP_V), CLAMP_V);
        lds_cw[(rbase + r) * KK + c2] = v;
      }
    }
  }
  __syncthreads();

  // ---- cross MLP: silu(m_ij@cw1+cb1)@cw2+cb2, clamp -> lds_ccw ------------
  {
    v8f acc = {};
#pragma unroll
    for (int ks = 0; ks < 2; ++ks) {
      v16h a = frag_A(lds_mij, HID_S, ks * 32, lane);
      v16h b = frag_B_swz(cw1h, 2, ks, wv, lane);
      acc = WMMA16(a, b, acc);
    }
    float bias = cb1[col];
#pragma unroll
    for (int r = 0; r < 8; ++r)
      lds_h2[(rbase + r) * HID_S + col] = (_Float16)silu_f(acc[r] + bias);
  }
  __syncthreads();
  {
#pragma unroll
    for (int nt = 0; nt < 2; ++nt) {
      int nTile = wv + nt * 4;
      v8f acc = {};
#pragma unroll
      for (int ks = 0; ks < 2; ++ks) {
        v16h a = frag_A(lds_h2, HID_S, ks * 32, lane);
        v16h b = frag_B_swz(cw2h, 2, ks, nTile, lane);
        acc = WMMA16(a, b, acc);
      }
      int c2 = nTile * 16 + (lane & 15);
      float bias = cb2[c2];
#pragma unroll
      for (int r = 0; r < 8; ++r) {
        float v = fminf(fmaxf(acc[r] + bias, -CLAMP_V), CLAMP_V);
        lds_ccw[(rbase + r) * KK + c2] = v;
      }
    }
  }
  __syncthreads();

  // ---- geometry + scatter: radial & cross-product updates (4 k per iter) --
  for (int it = tid * 4; it < 16 * KK; it += 128 * 4) {
    int e = it >> 7, k = it & 127;
    int s = s_idx[e], t = t_idx[e];
    if (s < 0) continue;

    v4f r0 = *(const v4f*)&lds_rel[(e * 3 + 0) * KK + k];
    v4f r1 = *(const v4f*)&lds_rel[(e * 3 + 1) * KK + k];
    v4f r2 = *(const v4f*)&lds_rel[(e * 3 + 2) * KK + k];
    v4f cwv = *(const v4f*)&lds_cw[e * KK + k];
    v4f ccv = *(const v4f*)&lds_ccw[e * KK + k];

    float invc_s = (s < BB) ? 1.0f / fmaxf(cnt[s], 1.0f) : 0.0f;
    float invc_t = (t < BB) ? 1.0f / fmaxf(cnt[t], 1.0f) : 0.0f;
    v4f xs[3], xt[3];
#pragma unroll
    for (int d = 0; d < 3; ++d) {
      v4f a = *(const v4f*)(Xn + ((long long)s * 3 + d) * KK + k);
      v4f b = *(const v4f*)(Xn + ((long long)t * 3 + d) * KK + k);
      v4f ms = {0, 0, 0, 0}, mt = {0, 0, 0, 0};
      if (s < BB) {
        v4f m = *(const v4f*)&meanXsum[(s * 3 + d) * KK + k];
#pragma unroll
        for (int j = 0; j < 4; ++j) ms[j] = m[j] * invc_s;
      }
      if (t < BB) {
        v4f m = *(const v4f*)&meanXsum[(t * 3 + d) * KK + k];
#pragma unroll
        for (int j = 0; j < 4; ++j) mt[j] = m[j] * invc_t;
      }
#pragma unroll
      for (int j = 0; j < 4; ++j) { xs[d][j] = a[j] - ms[j]; xt[d][j] = b[j] - mt[j]; }
    }

    float* od = out + ((long long)t * 3) * KK + k;
#pragma unroll
    for (int j = 0; j < 4; ++j) {
      float dist = r0[j] * r0[j] + r1[j] * r1[j] + r2[j] * r2[j];
      float inv = 1.0f / (1.0f + sqrtf(dist + 1e-8f));
      float wc = cwv[j] * inv;

      float c0 = xs[1][j] * xt[2][j] - xs[2][j] * xt[1][j];
      float c1 = xs[2][j] * xt[0][j] - xs[0][j] * xt[2][j];
      float c2 = xs[0][j] * xt[1][j] - xs[1][j] * xt[0][j];
      float cn = sqrtf(c0 * c0 + c1 * c1 + c2 * c2);
      float wcc = ccv[j] / (1.0f + cn);

      atomicAdd(od + 0 * KK + j, r0[j] * wc + c0 * wcc);
      atomicAdd(od + 1 * KK + j, r1[j] * wc + c1 * wcc);
      atomicAdd(od + 2 * KK + j, r2[j] * wc + c2 * wcc);
    }
  }
}

// -------------------- host-side launch --------------------------------------

extern "C" void kernel_launch(void* const* d_in, const int* in_sizes, int n_in,
                              void* d_out, int out_size, void* d_ws, size_t ws_size,
                              hipStream_t stream) {
  const int*   batch = (const int*)d_in[0];
  const float* X     = (const float*)d_in[1];
  const float* H     = (const float*)d_in[2];
  const int*   eidx  = (const int*)d_in[3];
  const float* te    = (const float*)d_in[4];
  const float* e3w   = (const float*)d_in[5];
  const float* lnw   = (const float*)d_in[6];
  const float* lnb   = (const float*)d_in[7];
  const float* mw1 = (const float*)d_in[8];   const float* mb1 = (const float*)d_in[9];
  const float* mw2 = (const float*)d_in[10];  const float* mb2 = (const float*)d_in[11];
  const float* xw1 = (const float*)d_in[12];  const float* xb1 = (const float*)d_in[13];
  const float* xw2 = (const float*)d_in[14];  const float* xb2 = (const float*)d_in[15];
  const float* cw1 = (const float*)d_in[16];  const float* cb1 = (const float*)d_in[17];
  const float* cw2 = (const float*)d_in[18];  const float* cb2 = (const float*)d_in[19];

  const int n  = in_sizes[0];         // N
  const int ne = in_sizes[3] / 2;     // E
  float* out = (float*)d_out;

  // ---- workspace layout (all chunks are 32B multiples) ----
  char* ws = (char*)d_ws;
  float* sumX     = (float*)ws;                       ws += sizeof(float) * BB * 384;
  float* cnt      = (float*)ws;                       ws += sizeof(float) * BB;
  float* normsum  = (float*)ws;                       ws += sizeof(float) * BB * KK;
  float* meanXsum = (float*)ws;                       ws += sizeof(float) * BB * 384;
  size_t zbytes = (size_t)((char*)ws - (char*)d_ws);
  float* Xn = (float*)ws;                             ws += sizeof(float) * (size_t)n * 384;
  float* Hn = (float*)ws;                             ws += sizeof(float) * (size_t)n * DD;
  _Float16* mw1h = (_Float16*)ws;                     ws += sizeof(_Float16) * MSG_IN * DD;
  _Float16* mw2h = (_Float16*)ws;                     ws += sizeof(_Float16) * DD * DD;
  _Float16* xw1h = (_Float16*)ws;                     ws += sizeof(_Float16) * DD * DD;
  _Float16* xw2h = (_Float16*)ws;                     ws += sizeof(_Float16) * DD * KK;
  _Float16* cw1h = (_Float16*)ws;                     ws += sizeof(_Float16) * DD * DD;
  _Float16* cw2h = (_Float16*)ws;                     ws += sizeof(_Float16) * DD * KK;

  hipMemsetAsync(d_ws, 0, zbytes, stream);

  swz_w_kernel<<<(MSG_IN * DD + 255) / 256, 256, 0, stream>>>(mw1, mw1h, MSG_IN, DD);
  swz_w_kernel<<<(DD * DD + 255) / 256, 256, 0, stream>>>(mw2, mw2h, DD, DD);
  swz_w_kernel<<<(DD * DD + 255) / 256, 256, 0, stream>>>(xw1, xw1h, DD, DD);
  swz_w_kernel<<<(DD * KK + 255) / 256, 256, 0, stream>>>(xw2, xw2h, DD, KK);
  swz_w_kernel<<<(DD * DD + 255) / 256, 256, 0, stream>>>(cw1, cw1h, DD, DD);
  swz_w_kernel<<<(DD * KK + 255) / 256, 256, 0, stream>>>(cw2, cw2h, DD, KK);

  long long tot1 = (long long)n * 384;
  accum_mean_kernel<<<(unsigned)((tot1 + 255) / 256), 256, 0, stream>>>(X, batch, sumX, cnt, n);

  long long tot2 = (long long)n * KK;
  center_norm_kernel<<<(unsigned)((tot2 + 255) / 256), 256, 0, stream>>>(
      X, batch, sumX, cnt, Xn, normsum, n);

  e3norm_kernel<<<(unsigned)((tot1 + 255) / 256), 256, 0, stream>>>(
      batch, e3w, normsum, cnt, Xn, out, meanXsum, n);

  layernorm_kernel<<<(n + 255) / 256, 256, 0, stream>>>(H, lnw, lnb, Hn, n);

  edge_kernel<<<(ne + 15) / 16, 128, 0, stream>>>(
      Xn, Hn, te, batch, eidx, cnt, meanXsum,
      mw1h, mw2h, xw1h, xw2h, cw1h, cw2h,
      mb1, mb2, xb1, xb2, cb1, cb2, out, ne);
}
